// DirectEncodingModel_28235115004379
// MI455X (gfx1250) — compile-verified, matrix operand order checked by reading, not run on hardware
//
#include <hip/hip_runtime.h>
#include <cmath>

// ---------------------------------------------------------------------------
// DirectEncodingModel on gfx1250:
//   4 levels of: gather(flat, idx) -> per-group 16x32 @ 32x16 GEMM -> +bias -> tanh
// One wave owns a 16-row batch tile; its growing "flat" feature vector
// (<=1024 f16) lives in LDS; each (tile, group) is ONE v_wmma_f32_16x16x32_f16
// (K = F = 32 exactly, f32 accumulate). Bias is pre-loaded into the WMMA C
// accumulator; tanh is the gfx1250 hardware v_tanh_f32 transcendental; gather
// addresses are fully pre-baked per lane (row base folded in by the prep
// kernel) so each gather is one ds_load_u16 + one dual-issuable shift.
// ---------------------------------------------------------------------------

typedef __attribute__((ext_vector_type(16))) _Float16    v16h;
typedef __attribute__((ext_vector_type(4)))  _Float16    v4h;
typedef __attribute__((ext_vector_type(8)))  float       v8f;
typedef __attribute__((ext_vector_type(4)))  float       v4f;
typedef __attribute__((ext_vector_type(4)))  int         v4i;
typedef __attribute__((ext_vector_type(4)))  unsigned int v4u;

#define B_TOTAL  65536
#define N_IN     256
#define G        16
#define U        16
#define F        32
#define LEVELS   4
#define ROWS     16        // batch rows per wave / workgroup
#define STRIDE_H 1032      // padded LDS row stride in f16 (2064 B = 516 dwords,
                           // 516 % 64 = 4 -> 16 rows land in 16 distinct banks)

// gfx1250 hardware tanh transcendental (ISA: TRANS ops incl. TANH).
#if __has_builtin(__builtin_amdgcn_tanhf)
#define DEV_TANH(x) __builtin_amdgcn_tanhf(x)
#else
#define DEV_TANH(x) tanhf(x)
#endif

// Per-lane K order of the 16-bit 16x32 A operand (ISA 7.12.2):
//   lanes 0-15 : K = {0..7, 16..23};  lanes 16-31 : K = {8..15, 24..31}
__device__ __forceinline__ int kmap(int half, int e) {
    return ((e < 8) ? (half ? 8 : 0) : (half ? 24 : 16)) + (e & 7);
}

// ---------------------------------------------------------------------------
// Prep: swizzle K[l] (G,F,U) into per-lane WMMA B-operand layout (f16 packed
// dwords) and idx[l] (G,F) into per-lane A-operand K order with the lane's
// LDS row base ((lane&15)*STRIDE_H) already folded in. Output in d_ws:
//   Bp[(l*G+g)*32 + lane][8]  dwords  (64 KB total)
//   Ip[(l*G+g)*32 + lane][16] ints    (128 KB total)
// ---------------------------------------------------------------------------
__global__ __launch_bounds__(32) void prep_kernel(
    const float* __restrict__ k1, const float* __restrict__ k2,
    const float* __restrict__ k3, const float* __restrict__ k4,
    const int*   __restrict__ i1, const int*   __restrict__ i2,
    const int*   __restrict__ i3, const int*   __restrict__ i4,
    unsigned int* __restrict__ Bp, int* __restrict__ Ip)
{
    const int lane = threadIdx.x;
    const int g    = blockIdx.x & (G - 1);
    const int l    = blockIdx.x >> 4;
    const float* K = (l == 0) ? k1 : (l == 1) ? k2 : (l == 2) ? k3 : k4;
    const int*   I = (l == 0) ? i1 : (l == 1) ? i2 : (l == 2) ? i3 : i4;
    const int n    = lane & 15;      // B-operand column (unit u) == A gather row
    const int half = lane >> 4;

    unsigned int bw[8];
    int          offs[16];
#pragma unroll
    for (int e = 0; e < 16; ++e) {
        const int k = kmap(half, e);
        const _Float16 h = (_Float16)K[(g * F + k) * U + n];
        const unsigned int hu = (unsigned int)__builtin_bit_cast(unsigned short, h);
        if (e & 1) bw[e >> 1] |= hu << 16;
        else       bw[e >> 1]  = hu;
        // Bake the lane's LDS row base into the gather index (element units).
        offs[e] = n * STRIDE_H + I[g * F + k];
    }
    unsigned int* bdst = Bp + (((size_t)(l * G + g)) * 32 + lane) * 8;
    *(v4u*)(bdst)     = (v4u){bw[0], bw[1], bw[2], bw[3]};
    *(v4u*)(bdst + 4) = (v4u){bw[4], bw[5], bw[6], bw[7]};
    int* idst = Ip + (((size_t)(l * G + g)) * 32 + lane) * 16;
#pragma unroll
    for (int j = 0; j < 4; ++j)
        *(v4i*)(idst + 4 * j) =
            (v4i){offs[4*j], offs[4*j+1], offs[4*j+2], offs[4*j+3]};
}

// ---------------------------------------------------------------------------
// Main: one wave per 16-row tile, flat kept in LDS (f16, bank-skewed stride).
// Same-wave DS ops are in-order on CDNA5 -> no barriers needed anywhere.
// ---------------------------------------------------------------------------
__global__ __launch_bounds__(32) void model_kernel(
    const float* __restrict__ x,
    const float* __restrict__ b1, const float* __restrict__ b2,
    const float* __restrict__ b3, const float* __restrict__ b4,
    const unsigned int* __restrict__ Bp, const int* __restrict__ Ip,
    float* __restrict__ out)
{
    __shared__ _Float16 flat[ROWS * STRIDE_H];   // 33,024 B
    const int  lane    = threadIdx.x;
    const long rowbase = (long)blockIdx.x * ROWS;

    // ---- stage x tile into LDS as f16 (coalesced b128 reads, b64 DS writes)
#pragma unroll 4
    for (int r = 0; r < ROWS; ++r) {
        const float* xr = x + (rowbase + r) * N_IN;
#pragma unroll
        for (int j = 0; j < 2; ++j) {
            const int c4 = lane + 32 * j;             // float4 index 0..63
            v4f v = *(const v4f*)(xr + 4 * c4);
            v4h h = {(_Float16)v.x, (_Float16)v.y, (_Float16)v.z, (_Float16)v.w};
            *(v4h*)&flat[r * STRIDE_H + 4 * c4] = h;
        }
    }

    const int n    = lane & 15;          // C/D column
    const int mrow = (lane >> 4) * 8;    // C/D row offset of upper half-wave
    const float* biases[LEVELS] = {b1, b2, b3, b4};

#pragma unroll
    for (int l = 0; l < LEVELS; ++l) {
        const float* __restrict__ bias = biases[l];
        const int opos = N_IN + l * (G * U);
        for (int g = 0; g < G; ++g) {
            // B operand: pre-swizzled f16, 2x coalesced b128
            const unsigned int* bp = Bp + (((size_t)(l * G + g)) * 32 + lane) * 8;
            struct B8 { v4u lo, hi; } braw = {*(const v4u*)bp, *(const v4u*)(bp + 4)};
            v16h bmat = __builtin_bit_cast(v16h, braw);
            // gather indices: per-lane K order, row base pre-baked: 4x b128
            const int* ip = Ip + (((size_t)(l * G + g)) * 32 + lane) * 16;
            v4i o0 = *(const v4i*)(ip);
            v4i o1 = *(const v4i*)(ip + 4);
            v4i o2 = *(const v4i*)(ip + 8);
            v4i o3 = *(const v4i*)(ip + 12);

            // Bias folded into the C accumulator: C layout is (M=r / r+8,
            // N = lane&15) in each of the 8 regs -> bias[g][n] broadcast.
            const float bv = bias[g * U + n];
            v8f acc = {bv, bv, bv, bv, bv, bv, bv, bv};

            // A operand: 16 ds_load_u16 gathers, zero address arithmetic
            // beyond the element->byte shift.
            v16h a;
            a[0]  = flat[o0.x];  a[1]  = flat[o0.y];
            a[2]  = flat[o0.z];  a[3]  = flat[o0.w];
            a[4]  = flat[o1.x];  a[5]  = flat[o1.y];
            a[6]  = flat[o1.z];  a[7]  = flat[o1.w];
            a[8]  = flat[o2.x];  a[9]  = flat[o2.y];
            a[10] = flat[o2.z];  a[11] = flat[o2.w];
            a[12] = flat[o3.x];  a[13] = flat[o3.y];
            a[14] = flat[o3.z];  a[15] = flat[o3.w];

            v8f d = __builtin_amdgcn_wmma_f32_16x16x32_f16(
                false, a, false, bmat, (short)0, acc, false, false);

            if (l == LEVELS - 1) {
                // final level: tanh -> f32 straight to HBM (coalesced 64B runs)
                float* op = out + (rowbase + mrow) * (G * U) + g * U + n;
#pragma unroll
                for (int r = 0; r < 8; ++r)
                    op[r * (G * U)] = DEV_TANH(d[r]);
            } else {
                // append tanh(out) to flat for the next level's gathers
#pragma unroll
                for (int r = 0; r < 8; ++r)
                    flat[(mrow + r) * STRIDE_H + opos + g * U + n] =
                        (_Float16)DEV_TANH(d[r]);
            }
        }
    }
}

// ---------------------------------------------------------------------------
// Input order (setup_inputs): x, k1,b1,idx1, k2,b2,idx2, k3,b3,idx3, k4,b4,idx4
// d_ws layout: [0,64KB) Bp swizzled weights, [64KB,192KB) Ip swizzled indices
// ---------------------------------------------------------------------------
extern "C" void kernel_launch(void* const* d_in, const int* in_sizes, int n_in,
                              void* d_out, int out_size, void* d_ws, size_t ws_size,
                              hipStream_t stream) {
    const float* x  = (const float*)d_in[0];
    const float* k1 = (const float*)d_in[1];
    const float* b1 = (const float*)d_in[2];
    const int*   i1 = (const int*)  d_in[3];
    const float* k2 = (const float*)d_in[4];
    const float* b2 = (const float*)d_in[5];
    const int*   i2 = (const int*)  d_in[6];
    const float* k3 = (const float*)d_in[7];
    const float* b3 = (const float*)d_in[8];
    const int*   i3 = (const int*)  d_in[9];
    const float* k4 = (const float*)d_in[10];
    const float* b4 = (const float*)d_in[11];
    const int*   i4 = (const int*)  d_in[12];

    unsigned int* Bp = (unsigned int*)d_ws;
    int*          Ip = (int*)((char*)d_ws + 64 * 1024);

    prep_kernel<<<dim3(LEVELS * G), dim3(32), 0, stream>>>(
        k1, k2, k3, k4, i1, i2, i3, i4, Bp, Ip);
    model_kernel<<<dim3(B_TOTAL / ROWS), dim3(32), 0, stream>>>(
        x, b1, b2, b3, b4, Bp, Ip, (float*)d_out);
}
